// GCN_24970939859192
// MI455X (gfx1250) — matrix operand order repaired and checked by reading, not compiled, not run on hardware
//
#include <hip/hip_runtime.h>
#include <hip/hip_bf16.h>

#define IN_DIM_C  1433
#define HIDDEN_C  128
#define OUT_DIM_C 7

typedef __attribute__((ext_vector_type(16))) __bf16        v16bf;
typedef __attribute__((ext_vector_type(8)))  float         v8f;
typedef __attribute__((ext_vector_type(4)))  unsigned int  v4u;

struct U8x32 { v4u lo, hi; };   // 32 bytes, bit_cast target for v16bf

__device__ __forceinline__ unsigned int pack_bf16x2(float a, float b) {
    __bf16 x = (__bf16)a, y = (__bf16)b;
    unsigned short lo = __builtin_bit_cast(unsigned short, x);
    unsigned short hi = __builtin_bit_cast(unsigned short, y);
    return (unsigned int)lo | ((unsigned int)hi << 16);
}

// ---------------------------------------------------------------------------
// Degree / normalization kernels
// ---------------------------------------------------------------------------
__global__ void deg_init_k(float* __restrict__ deg, int n_nodes) {
    int i = blockIdx.x * blockDim.x + threadIdx.x;
    if (i < n_nodes) deg[i] = 1.0f;  // self loop contributes 1
}

__global__ void deg_count_k(const int* __restrict__ dst, float* __restrict__ deg,
                            int n_edges) {
    int e = blockIdx.x * blockDim.x + threadIdx.x;
    if (e < n_edges) atomicAdd(&deg[dst[e]], 1.0f);
}

__global__ void dinv_k(const float* __restrict__ deg, float* __restrict__ dinv,
                       int n_nodes) {
    int i = blockIdx.x * blockDim.x + threadIdx.x;
    if (i < n_nodes) {
        float d = deg[i];
        dinv[i] = (d > 0.0f) ? rsqrtf(d) : 0.0f;
    }
}

// ---------------------------------------------------------------------------
// Layer-1 GEMM: H = X @ W1  (100000 x 1433) @ (1433 x 128), bf16 WMMA, f32 acc
// Block = 256 threads = 8 wave32; block computes 128 rows x 128 cols.
// Each wave computes a 16x128 strip = 8 WMMA 16x16 tiles.
// LDS layouts are pre-swizzled so every fragment load is 2x ds_load_b128;
// B-fragment loads are software-pipelined one N-tile ahead of the WMMAs.
// ---------------------------------------------------------------------------
// A swizzle: per row (stride 20 uints = 80 B, 16B aligned, bank-staggered),
//   element order = K{0-31} contiguous (uint = packed K-pair)
//   -> lane-half h reads contiguous uints [h*8, h*8+8)
// B swizzle: per (ntile n, frag-lane l) a contiguous 8-uint run at
//   (n*32 + l)*12 (48 B stride, 16B aligned, bank-staggered);
//   run holds col = n*16 + (l&15), K = (l>>4)*16 + {0..15}
__global__ __launch_bounds__(256) void gcn_gemm1_wmma(
    const float* __restrict__ X, const float* __restrict__ W1,
    float* __restrict__ H, int n_nodes) {
    __shared__ unsigned int lA_u[128 * 20];      // 10.0 KB
    __shared__ unsigned int lB_u[8 * 32 * 12];   // 12.0 KB

    const int lane = threadIdx.x & 31;
    const int wv   = threadIdx.x >> 5;
    const int blockRow = blockIdx.x * 128;

    v8f acc[8];
#pragma unroll
    for (int n = 0; n < 8; ++n) { v8f z = {}; acc[n] = z; }

    const int h8   = lane >> 4;                       // lane half
    const int arow = wv * 16 + (lane & 15);           // A fragment row
    const unsigned int* aptr = &lA_u[arow * 20 + h8 * 8];

    for (int k0 = 0; k0 < IN_DIM_C; k0 += 32) {
        // ---- Stage A chunk: 128 rows x 32 K (fp32 -> packed bf16 pairs) ----
        for (int i = threadIdx.x; i < 128 * 16; i += 256) {
            int r = i >> 4;          // row in tile
            int t = i & 15;          // K-pair index (K = 2t, 2t+1)
            int gr = blockRow + r;
            int gk = k0 + 2 * t;
            float v0 = 0.0f, v1 = 0.0f;
            if (gr < n_nodes) {
                if (gk < IN_DIM_C)     v0 = X[(size_t)gr * IN_DIM_C + gk];
                if (gk + 1 < IN_DIM_C) v1 = X[(size_t)gr * IN_DIM_C + gk + 1];
            }
            int g = t >> 2;                          // K group of 8
            int slot = g;                            // identity: contiguous K
            lA_u[r * 20 + slot * 4 + (t & 3)] = pack_bf16x2(v0, v1);
        }
        // ---- Stage B chunk: 32 K x 128 cols (fp32 -> packed bf16 pairs) ----
        for (int i = threadIdx.x; i < 16 * 128; i += 256) {
            int c  = i & 127;        // column
            int kp = i >> 7;         // K-pair index (K = 2kp, 2kp+1)
            int gk = k0 + 2 * kp;
            float v0 = (gk < IN_DIM_C)     ? W1[(size_t)gk * HIDDEN_C + c]       : 0.0f;
            float v1 = (gk + 1 < IN_DIM_C) ? W1[(size_t)(gk + 1) * HIDDEN_C + c] : 0.0f;
            int n      = c >> 4;
            int flane  = (kp >> 3) * 16 + (c & 15);  // fragment lane
            lB_u[(n * 32 + flane) * 12 + (kp & 7)] = pack_bf16x2(v0, v1);
        }
        __syncthreads();

        // ---- A fragment: two b128 loads ----
        U8x32 at;
        at.lo = *(const v4u*)(aptr);
        at.hi = *(const v4u*)(aptr + 4);
        v16bf afrag = __builtin_bit_cast(v16bf, at);

        // ---- 8 N-tiles, B loads pipelined one tile ahead of the WMMA ----
        const unsigned int* bp0 = &lB_u[lane * 12];
        U8x32 bt;
        bt.lo = *(const v4u*)(bp0);
        bt.hi = *(const v4u*)(bp0 + 4);
#pragma unroll
        for (int n = 0; n < 8; ++n) {
            v16bf bfrag = __builtin_bit_cast(v16bf, bt);
            if (n < 7) {
                const unsigned int* bptr = &lB_u[((n + 1) * 32 + lane) * 12];
                bt.lo = *(const v4u*)(bptr);
                bt.hi = *(const v4u*)(bptr + 4);
            }
            acc[n] = __builtin_amdgcn_wmma_f32_16x16x32_bf16(
                false, afrag, false, bfrag, (short)0, acc[n], false, false);
        }
        __syncthreads();
    }

    // C/D layout: lane<16 -> rows M=0..7 in v0..7; lane>=16 -> rows M=8..15.
    const int row0 = blockRow + wv * 16 + h8 * 8;
    const int colL = lane & 15;
#pragma unroll
    for (int n = 0; n < 8; ++n) {
#pragma unroll
        for (int j = 0; j < 8; ++j) {
            int r = row0 + j;
            if (r < n_nodes)
                H[(size_t)r * HIDDEN_C + n * 16 + colL] = acc[n][j];
        }
    }
}

// ---------------------------------------------------------------------------
// Layer-1 edge scatter: out1[dst] += H[src] * dinv[src]*dinv[dst]
// One thread per (edge, 4 features): float4 gather, 4 f32 atomics (L2-resident)
// ---------------------------------------------------------------------------
__global__ void scatter1_k(const int* __restrict__ src, const int* __restrict__ dst,
                           const float* __restrict__ dinv, const float* __restrict__ H,
                           float* __restrict__ out1, int n_edges) {
    long long tid = (long long)blockIdx.x * blockDim.x + threadIdx.x;
    if (tid >= (long long)n_edges * 32) return;
    int e  = (int)(tid >> 5);
    int f4 = (int)(tid & 31) * 4;
    int s = src[e], d = dst[e];
    float w = dinv[s] * dinv[d];
    const float4 hv = *(const float4*)(H + (size_t)s * HIDDEN_C + f4);
    float* o = out1 + (size_t)d * HIDDEN_C + f4;
    atomicAdd(o + 0, hv.x * w);
    atomicAdd(o + 1, hv.y * w);
    atomicAdd(o + 2, hv.z * w);
    atomicAdd(o + 3, hv.w * w);
}

// out1 = relu(out1 + H*dinv^2 + b1)  (self loop + bias + activation, in place)
__global__ void epilogue1_k(float* __restrict__ out1, const float* __restrict__ H,
                            const float* __restrict__ dinv, const float* __restrict__ b1,
                            int n_nodes) {
    int tid = blockIdx.x * blockDim.x + threadIdx.x;
    if (tid >= n_nodes * HIDDEN_C) return;
    int i = tid >> 7, f = tid & 127;
    float di = dinv[i];
    float v = out1[tid] + H[tid] * di * di + b1[f];
    out1[tid] = v > 0.0f ? v : 0.0f;
}

// ---------------------------------------------------------------------------
// Layer-2 GEMM: H2 = out1 @ W2  (100000 x 128) @ (128 x 7) -- tiny, scalar FMA
// ---------------------------------------------------------------------------
__global__ void gemm2_k(const float* __restrict__ Hg, const float* __restrict__ W2,
                        float* __restrict__ H2, int n_nodes) {
    int tid = blockIdx.x * blockDim.x + threadIdx.x;
    if (tid >= n_nodes * 8) return;
    int i = tid >> 3, f = tid & 7;
    if (f >= OUT_DIM_C) return;
    const float* hr = Hg + (size_t)i * HIDDEN_C;
    float acc = 0.0f;
#pragma unroll 8
    for (int k = 0; k < HIDDEN_C; ++k) acc += hr[k] * W2[k * OUT_DIM_C + f];
    H2[(size_t)i * OUT_DIM_C + f] = acc;
}

__global__ void scatter2_k(const int* __restrict__ src, const int* __restrict__ dst,
                           const float* __restrict__ dinv, const float* __restrict__ H2,
                           float* __restrict__ out, int n_edges) {
    long long tid = (long long)blockIdx.x * blockDim.x + threadIdx.x;
    if (tid >= (long long)n_edges * 8) return;
    int e = (int)(tid >> 3);
    int f = (int)(tid & 7);
    if (f >= OUT_DIM_C) return;
    int s = src[e], d = dst[e];
    float w = dinv[s] * dinv[d];
    atomicAdd(out + (size_t)d * OUT_DIM_C + f, H2[(size_t)s * OUT_DIM_C + f] * w);
}

// out += H2*dinv^2 + b2  (self loop + bias, in place)
__global__ void epilogue2_k(float* __restrict__ out, const float* __restrict__ H2,
                            const float* __restrict__ dinv, const float* __restrict__ b2,
                            int n_nodes) {
    int tid = blockIdx.x * blockDim.x + threadIdx.x;
    if (tid >= n_nodes * OUT_DIM_C) return;
    int i = tid / OUT_DIM_C;
    int f = tid - i * OUT_DIM_C;
    float di = dinv[i];
    out[tid] += H2[tid] * di * di + b2[f];
}

// ---------------------------------------------------------------------------
// Launcher
// ---------------------------------------------------------------------------
extern "C" void kernel_launch(void* const* d_in, const int* in_sizes, int n_in,
                              void* d_out, int out_size, void* d_ws, size_t ws_size,
                              hipStream_t stream) {
    const float* X  = (const float*)d_in[0];
    const int*   ei = (const int*)d_in[1];
    const float* W1 = (const float*)d_in[2];
    const float* b1 = (const float*)d_in[3];
    const float* W2 = (const float*)d_in[4];
    const float* b2 = (const float*)d_in[5];
    float* out = (float*)d_out;

    const int n_nodes = in_sizes[0] / IN_DIM_C;
    const int n_edges = in_sizes[1] / 2;
    const int* src = ei;
    const int* dst = ei + n_edges;

    // Carve workspace (aligned 512B): deg, dinv, H, out1, H2  (~106 MB total)
    size_t off = 0;
    auto carve = [&](size_t bytes) {
        size_t p = off;
        off += (bytes + 511) & ~(size_t)511;
        return p;
    };
    char* wsb = (char*)d_ws;
    float* deg  = (float*)(wsb + carve((size_t)n_nodes * sizeof(float)));
    float* dinv = (float*)(wsb + carve((size_t)n_nodes * sizeof(float)));
    float* H    = (float*)(wsb + carve((size_t)n_nodes * HIDDEN_C * sizeof(float)));
    float* out1 = (float*)(wsb + carve((size_t)n_nodes * HIDDEN_C * sizeof(float)));
    float* H2   = (float*)(wsb + carve((size_t)n_nodes * OUT_DIM_C * sizeof(float)));
    (void)ws_size; (void)n_in; (void)out_size;

    // Zero accumulation targets (graph-capture-safe async memsets)
    hipMemsetAsync(out1, 0, (size_t)n_nodes * HIDDEN_C * sizeof(float), stream);
    hipMemsetAsync(out,  0, (size_t)n_nodes * OUT_DIM_C * sizeof(float), stream);

    // Degrees + symmetric normalization
    deg_init_k<<<(n_nodes + 255) / 256, 256, 0, stream>>>(deg, n_nodes);
    deg_count_k<<<(n_edges + 255) / 256, 256, 0, stream>>>(dst, deg, n_edges);
    dinv_k<<<(n_nodes + 255) / 256, 256, 0, stream>>>(deg, dinv, n_nodes);

    // Layer 1: H = X@W1 (WMMA), aggregate, +b1, relu
    gcn_gemm1_wmma<<<(n_nodes + 127) / 128, 256, 0, stream>>>(X, W1, H, n_nodes);
    {
        long long t = (long long)n_edges * 32;
        scatter1_k<<<(unsigned)((t + 255) / 256), 256, 0, stream>>>(
            src, dst, dinv, H, out1, n_edges);
    }
    epilogue1_k<<<(n_nodes * HIDDEN_C + 255) / 256, 256, 0, stream>>>(
        out1, H, dinv, b1, n_nodes);

    // Layer 2: H2 = out1@W2, aggregate into d_out, +b2
    gemm2_k<<<(n_nodes * 8 + 255) / 256, 256, 0, stream>>>(out1, W2, H2, n_nodes);
    {
        long long t = (long long)n_edges * 8;
        scatter2_k<<<(unsigned)((t + 255) / 256), 256, 0, stream>>>(
            src, dst, dinv, H2, out, n_edges);
    }
    epilogue2_k<<<(n_nodes * OUT_DIM_C + 255) / 256, 256, 0, stream>>>(
        out, H2, dinv, b2, n_nodes);
}
